// KernelSplattingInterpolator_34900904247903
// MI455X (gfx1250) — matrix operand order repaired and checked by reading, not compiled, not run on hardware
//
#include <hip/hip_runtime.h>

typedef __attribute__((ext_vector_type(16))) _Float16 v16h;
typedef __attribute__((ext_vector_type(8)))  _Float16 v8h;
typedef __attribute__((ext_vector_type(8)))  float    v8f;

#define QTOT 16384
#define NTOT 8192
#define CCH  64
#define KC   32          // K-chunk (data points per WMMA step)
#define WAVES 8          // waves per block, one 16-query tile per wave
#define BQ   (WAVES * 16)
#define BPAD 24          // halfs per B-staging row (16 data + 8 pad -> 48B, bank-friendly)

__global__ __launch_bounds__(256) void
splat_wmma_kernel(const float* __restrict__ q_pos,
                  const float* __restrict__ d_pos,
                  const float* __restrict__ d_val,
                  float* __restrict__ out)
{
    __shared__ float4   s_dpos[KC];                // d_pos chunk, xyz + pad
    __shared__ _Float16 s_dvalB[4][32][BPAD];      // [c-tile][lane][element] B-operand layout

    const int tid  = threadIdx.x;
    const int lane = tid & 31;
    const int wave = tid >> 5;
    const int m    = lane & 15;          // output row within tile / B column
    const bool hi  = lane >= 16;
    const int kbase = hi ? 8 : 0;        // A-operand K base for this half-wave

    // per-lane query position (row m of this wave's 16-query tile)
    const int qrow = blockIdx.x * BQ + wave * 16 + m;
    const float qx = q_pos[qrow * 3 + 0];
    const float qy = q_pos[qrow * 3 + 1];
    const float qz = q_pos[qrow * 3 + 2];

    v8f acc0 = {}, acc1 = {}, acc2 = {}, acc3 = {};
    float swsum = 0.0f;
    // exp(-r2/(2*sigma^2)) = exp2(r2 * (-50 * log2(e)))
    const float SC = -50.0f * 1.44269504088896340736f;

    for (int n0 = 0; n0 < NTOT; n0 += KC) {
        __syncthreads();   // previous chunk's compute done before LDS overwrite

        // ---- stage d_pos chunk: 32 points -> float4 slots (128 floats) ----
        if (tid < 128) {
            const int p = tid >> 2, comp = tid & 3;
            ((float*)s_dpos)[tid] = (comp < 3) ? d_pos[(n0 + p) * 3 + comp] : 0.0f;
        }

        // ---- stage d_val chunk (32 x 64 f32) into f16 B-operand layout ----
        #pragma unroll
        for (int r = 0; r < 2; ++r) {
            const int idx = r * 1024 + tid * 4;          // flat idx in 32x64 chunk
            const int k = idx >> 6, c = idx & 63;        // data-point row, channel
            const float4 dv = *(const float4*)(d_val + (size_t)(n0 + k) * CCH + c);
            const int e   = (k & 7) | ((k >> 4) << 3);   // B element index
            const int klo = ((k >> 3) & 1) << 4;         // +16 lanes for K in 8-15, 24-31
            const int t   = c >> 4;                      // channel tile
            const int ln  = c & 15;                      // B column within tile
            s_dvalB[t][klo + ln + 0][e] = (_Float16)dv.x;
            s_dvalB[t][klo + ln + 1][e] = (_Float16)dv.y;
            s_dvalB[t][klo + ln + 2][e] = (_Float16)dv.z;
            s_dvalB[t][klo + ln + 3][e] = (_Float16)dv.w;
        }

        // prefetch next chunk of d_val into cache (32B per thread covers 8KB)
        if (n0 + KC < NTOT)
            __builtin_prefetch(d_val + (size_t)(n0 + KC) * CCH + tid * 8, 0, 1);

        __syncthreads();

        // ---- build A operand: 16x32 weight tile, computed in-register ----
        v16h a;
        #pragma unroll
        for (int e = 0; e < 16; ++e) {
            const int k = kbase + (e & 7) + ((e & 8) << 1);   // A-layout K index
            const float4 dp = s_dpos[k];
            const float dx = qx - dp.x, dy = qy - dp.y, dz = qz - dp.z;
            const float r2 = dx * dx + dy * dy + dz * dz;
            const float w  = __builtin_amdgcn_exp2f(r2 * SC); // v_exp_f32
            swsum += w;                                       // f32 row-sum (free)
            a[e] = (_Float16)w;
        }

        // ---- load four 32x16 B tiles from LDS (already in operand layout) ----
        #pragma unroll
        for (int t = 0; t < 4; ++t) {
            const v8h blo = *(const v8h*)&s_dvalB[t][lane][0];
            const v8h bhi = *(const v8h*)&s_dvalB[t][lane][8];
            const v16h b = __builtin_shufflevector(blo, bhi,
                0, 1, 2, 3, 4, 5, 6, 7, 8, 9, 10, 11, 12, 13, 14, 15);
            if (t == 0)
                acc0 = __builtin_amdgcn_wmma_f32_16x16x32_f16(false, a, false, b,
                                                              (short)0, acc0, false, false);
            else if (t == 1)
                acc1 = __builtin_amdgcn_wmma_f32_16x16x32_f16(false, a, false, b,
                                                              (short)0, acc1, false, false);
            else if (t == 2)
                acc2 = __builtin_amdgcn_wmma_f32_16x16x32_f16(false, a, false, b,
                                                              (short)0, acc2, false, false);
            else
                acc3 = __builtin_amdgcn_wmma_f32_16x16x32_f16(false, a, false, b,
                                                              (short)0, acc3, false, false);
        }
    }

    // ---- finalize: wsum per row, normalize, store ----
    // lane m holds K{0-7,16-23,...}, lane m+16 holds the complement
    const float wtot = swsum + __shfl_xor(swsum, 16, 32);  // full wsum[m] in all lanes

    // D layout: lane column n = m; VGPR j holds row (j + 8*hi)
    float* op = out + (size_t)(blockIdx.x * BQ + wave * 16) * CCH;
    #pragma unroll
    for (int j = 0; j < 8; ++j) {
        const int Mrow = j + (hi ? 8 : 0);
        const float ws = __shfl(wtot, Mrow, 32);           // wsum for this output row
        const float s  = 1.0f / (1e-4f + ws);
        op[(size_t)Mrow * CCH +  0 + m] = acc0[j] * s;
        op[(size_t)Mrow * CCH + 16 + m] = acc1[j] * s;
        op[(size_t)Mrow * CCH + 32 + m] = acc2[j] * s;
        op[(size_t)Mrow * CCH + 48 + m] = acc3[j] * s;
    }
}

extern "C" void kernel_launch(void* const* d_in, const int* in_sizes, int n_in,
                              void* d_out, int out_size, void* d_ws, size_t ws_size,
                              hipStream_t stream) {
    const float* q_pos = (const float*)d_in[0];
    const float* d_pos = (const float*)d_in[1];
    const float* d_val = (const float*)d_in[2];
    float* out = (float*)d_out;

    dim3 block(256);                 // 8 waves, one 16-query tile each
    dim3 grid(QTOT / BQ);            // 128 blocks
    hipLaunchKernelGGL(splat_wmma_kernel, grid, block, 0, stream,
                       q_pos, d_pos, d_val, out);
}
